// GCN_80247168959127
// MI455X (gfx1250) — matrix operand order repaired and checked by reading, not compiled, not run on hardware
//
#include <hip/hip_runtime.h>
#include <hip/hip_bf16.h>

typedef __attribute__((ext_vector_type(16))) _Float16 v16h;
typedef __attribute__((ext_vector_type(8)))  float    v8f;

// ---------------------------------------------------------------------------
// Degree: deg[c] += 1 for every edge endpoint in col[] (self-loop added later)
// ---------------------------------------------------------------------------
__global__ void GCN_deg_kernel(const long long* __restrict__ col,
                               float* __restrict__ deg, long long E) {
  long long t = (long long)blockIdx.x * blockDim.x + threadIdx.x;
  if (t < E) atomicAdd(&deg[(int)col[t]], 1.0f);
}

// dinv[n] = rsqrt(deg[n] + 1)   (+1 = self loop; strictly positive)
__global__ void GCN_dinv_kernel(float* __restrict__ deg, int N) {
  int t = blockIdx.x * blockDim.x + threadIdx.x;
  if (t < N) deg[t] = rsqrtf(deg[t] + 1.0f);
}

// ---------------------------------------------------------------------------
// Edge preprocessing (graph is reused by all 3 layers):
//   rc[e]  = (int32 row, int32 col)   -- halves index bytes vs int64
//   nrm[e] = dinv[row]*dinv[col]      -- hoists 2 gathers out of the hot loop
// ---------------------------------------------------------------------------
__global__ void GCN_edgeprep_kernel(const long long* __restrict__ rowi,
                                    const long long* __restrict__ coli,
                                    const float* __restrict__ dinv,
                                    int2* __restrict__ rc,
                                    float* __restrict__ nrm, long long E) {
  long long e = (long long)blockIdx.x * blockDim.x + threadIdx.x;
  if (e < E) {
    const int r = (int)rowi[e];
    const int c = (int)coli[e];
    rc[e] = make_int2(r, c);
    nrm[e] = dinv[r] * dinv[c];
  }
}

// ---------------------------------------------------------------------------
// Prepack W[64 x F] (f32, row-major) into the per-lane WMMA B-operand layout
// as f16.  Bp index = ((colTile*2 + kkIdx)*32 + lane)*16 + i  where half slot
// i of lane l holds K = kkIdx*32 + (i<8 ? i : i+8) + (l<16 ? 0 : 8), N-col = l&15
// (ISA 7.12.2, 16-bit B 32x16).  Total = (F/16)*1024 halves (<= 8 KB).
// ---------------------------------------------------------------------------
__global__ void GCN_packB_kernel(const float* __restrict__ W,
                                 _Float16* __restrict__ Bp, int F) {
  const int total = (F / 16) * 1024;
  for (int t = blockIdx.x * blockDim.x + threadIdx.x; t < total;
       t += gridDim.x * blockDim.x) {
    const int i       = t & 15;
    const int lane    = (t >> 4) & 31;
    const int chunk   = t >> 9;          // colTile*2 + kkIdx
    const int kkIdx   = chunk & 1;
    const int colTile = chunk >> 1;
    const int hi      = lane >> 4;
    const int n       = lane & 15;
    const int k       = kkIdx * 32 + hi * 8 + i + (i >> 3) * 8;
    Bp[t] = (_Float16)W[k * F + colTile * 16 + n];
  }
}

// ---------------------------------------------------------------------------
// H[N x F] = X[N x 64] @ W[64 x F] via V_WMMA_F32_16X16X32_F16.
// One wave per 16x16 tile; K=64 as two k32 steps.  F is a compile-time
// constant so the 8 D-stores share one base address + immediate offsets.
// A 16x32 f16 layout: lane l (row m=l&15) half-slot i holds
//   K = kk + (i<8 ? i : i+8) + (l<16?0:8) -> two contiguous runs of 8 K's
//   -> 2x float4 (b128) loads per run, packed with v_cvt_pk_f16_f32.
// B is prepacked: one 32-byte (2x b128) load per k-step.
// D 16x16 f32: VGPR j -> row (j + (l<16?0:8)), col = l&15.
// ---------------------------------------------------------------------------
template <int F>
__global__ void GCN_gemm_wmma_f16(const float* __restrict__ X,
                                  const _Float16* __restrict__ Bpack,
                                  float* __restrict__ H, int nRowTiles) {
  const int wave = threadIdx.x >> 5;
  const int lane = threadIdx.x & 31;
  const int rowTile = blockIdx.x * 4 + wave;   // wave-uniform: EXEC stays all-1s
  const int colTile = blockIdx.y;
  if (rowTile >= nRowTiles) return;

  const int m  = lane & 15;
  const int hi = lane >> 4;
  const int row = rowTile * 16 + m;
  const int col = colTile * 16 + m;

  const float4* __restrict__ xr = (const float4*)(X + (size_t)row * 64);
  const v16h*   __restrict__ bp = (const v16h*)Bpack;

  v8f acc = {};
#pragma unroll
  for (int kkIdx = 0; kkIdx < 2; ++kkIdx) {
    const int kq = kkIdx * 8;                  // kk/4
    const float4 a0 = xr[kq + hi * 2 + 0];
    const float4 a1 = xr[kq + hi * 2 + 1];
    const float4 a2 = xr[kq + 4 + hi * 2 + 0];
    const float4 a3 = xr[kq + 4 + hi * 2 + 1];
    v16h a;
    a[0]  = (_Float16)a0.x;  a[1]  = (_Float16)a0.y;
    a[2]  = (_Float16)a0.z;  a[3]  = (_Float16)a0.w;
    a[4]  = (_Float16)a1.x;  a[5]  = (_Float16)a1.y;
    a[6]  = (_Float16)a1.z;  a[7]  = (_Float16)a1.w;
    a[8]  = (_Float16)a2.x;  a[9]  = (_Float16)a2.y;
    a[10] = (_Float16)a2.z;  a[11] = (_Float16)a2.w;
    a[12] = (_Float16)a3.x;  a[13] = (_Float16)a3.y;
    a[14] = (_Float16)a3.z;  a[15] = (_Float16)a3.w;

    const v16h b = bp[(colTile * 2 + kkIdx) * 32 + lane];

    // (neg_a, A, neg_b, B, c_mod, C, reuse_a, reuse_b)
    acc = __builtin_amdgcn_wmma_f32_16x16x32_f16(false, a, false, b,
                                                 (short)0, acc, false, false);
  }

  float* __restrict__ hp = H + (size_t)(rowTile * 16 + hi * 8) * F + col;
#pragma unroll
  for (int j = 0; j < 8; ++j) hp[j * F] = acc[j];   // immediate offsets
}

// ---------------------------------------------------------------------------
// Edge scatter, 4 features per thread (b128 gather + 4 f32 atomics):
//   agg[c*F + 4q..4q+3] += h[r*F + 4q..4q+3] * nrm[e]
// FQ = F/4, LG = log2(FQ) (compile-time).
// ---------------------------------------------------------------------------
template <int FQ, int LG>
__global__ void GCN_scatter_kernel(const int2* __restrict__ rc,
                                   const float* __restrict__ nrm,
                                   const float4* __restrict__ h4,
                                   float* __restrict__ agg, long long E) {
  const int q = threadIdx.x & (FQ - 1);
  const long long e = (long long)blockIdx.x * (256 >> LG) + (threadIdx.x >> LG);
  if (e < E) {
    const int2 p = rc[e];
    const float nv = nrm[e];
    const float4 v = h4[(long long)p.x * FQ + q];
    float* dst = agg + ((long long)p.y * FQ + q) * 4;
    atomicAdd(dst + 0, v.x * nv);
    atomicAdd(dst + 1, v.y * nv);
    atomicAdd(dst + 2, v.z * nv);
    atomicAdd(dst + 3, v.w * nv);
  }
}

// ---------------------------------------------------------------------------
// out[n,f] = [relu]( agg[n,f] + g[n,f]*dinv[n]^2 + bias[f] ), float4-wide.
// (g*dinv^2 is the self-loop message, added without atomics)
// ---------------------------------------------------------------------------
template <int FQ, int LG>
__global__ void GCN_finalize_kernel(const float4* __restrict__ agg4,
                                    const float4* __restrict__ g4,
                                    const float* __restrict__ dinv,
                                    const float4* __restrict__ bias4,
                                    float4* __restrict__ out4,
                                    int N, int doRelu) {
  const long long t = (long long)blockIdx.x * blockDim.x + threadIdx.x;
  if (t < (long long)N * FQ) {
    const int n = (int)(t >> LG);
    const int q = (int)(t & (FQ - 1));
    const float di2 = dinv[n] * dinv[n];
    const float4 a = agg4[t];
    const float4 g = g4[t];
    const float4 b = bias4[q];
    float4 v;
    v.x = a.x + g.x * di2 + b.x;
    v.y = a.y + g.y * di2 + b.y;
    v.z = a.z + g.z * di2 + b.z;
    v.w = a.w + g.w * di2 + b.w;
    if (doRelu) {
      v.x = fmaxf(v.x, 0.0f); v.y = fmaxf(v.y, 0.0f);
      v.z = fmaxf(v.z, 0.0f); v.w = fmaxf(v.w, 0.0f);
    }
    out4[t] = v;
  }
}

extern "C" void kernel_launch(void* const* d_in, const int* in_sizes, int n_in,
                              void* d_out, int out_size, void* d_ws, size_t ws_size,
                              hipStream_t stream) {
  const float* x  = (const float*)d_in[0];   // [N,64]
  const float* W1 = (const float*)d_in[1];   // [64,64]
  const float* b1 = (const float*)d_in[2];   // [64]
  const float* W2 = (const float*)d_in[3];   // [64,64]
  const float* b2 = (const float*)d_in[4];   // [64]
  const float* Wo = (const float*)d_in[5];   // [64,16]
  const float* bo = (const float*)d_in[6];   // [16]
  const long long* ei = (const long long*)d_in[7];  // int64 [2,E]

  const long long E = (long long)in_sizes[7] / 2;
  const int N = in_sizes[0] / 64;
  const long long* rowi = ei;
  const long long* coli = ei + E;

  // Workspace layout (floats): dinv[N] | gbuf[N*64] | agg[N*64] |
  //   Bp1[4096h] Bp2[4096h] Bpo[1024h] | rc[2E int] | nrm[E]   (~35.5 MB)
  float* ws   = (float*)d_ws;
  float* dinv = ws;
  float* gbuf = ws + N;
  float* agg  = gbuf + (size_t)N * 64;
  _Float16* Bp1 = (_Float16*)(agg + (size_t)N * 64);
  _Float16* Bp2 = Bp1 + 4096;
  _Float16* Bpo = Bp2 + 4096;
  int2*  rc  = (int2*)(Bpo + 1024);          // 16B-aligned (half counts even)
  float* nrm = (float*)(rc + E);

  // Output: [ out (N*16) | pen (N*64) ]; pen region doubles as hidden buffer.
  float* out0 = (float*)d_out;
  float* pen  = out0 + (size_t)N * 16;

  const int eBlocks = (int)((E + 255) / 256);

  // --- normalization, edge prep, weight prepack ----------------------------
  hipMemsetAsync(dinv, 0, (size_t)N * sizeof(float), stream);
  GCN_deg_kernel<<<eBlocks, 256, 0, stream>>>(coli, dinv, E);
  GCN_dinv_kernel<<<(N + 255) / 256, 256, 0, stream>>>(dinv, N);
  GCN_edgeprep_kernel<<<eBlocks, 256, 0, stream>>>(rowi, coli, dinv, rc, nrm, E);
  GCN_packB_kernel<<<1, 256, 0, stream>>>(W1, Bp1, 64);
  GCN_packB_kernel<<<1, 256, 0, stream>>>(W2, Bp2, 64);
  GCN_packB_kernel<<<1, 256, 0, stream>>>(Wo, Bpo, 16);

  const int nRowTiles = N / 16;             // 50000 is a multiple of 16
  const dim3 g64((nRowTiles + 3) / 4, 4);   // F=64 -> 4 col tiles
  const dim3 g16((nRowTiles + 3) / 4, 1);   // F=16 -> 1 col tile
  const int fin64 = (int)(((long long)N * 16 + 255) / 256);  // N*64/4 float4s
  const int fin16 = (int)(((long long)N * 4  + 255) / 256);  // N*16/4 float4s
  const int sc64  = (int)((E + 15) / 16);   // 16 edges per block (FQ=16)
  const int sc16  = (int)((E + 63) / 64);   // 64 edges per block (FQ=4)

  // --- layer 1: h1 = relu(Agg(x@W1) + b1) -> pen region --------------------
  GCN_gemm_wmma_f16<64><<<g64, 128, 0, stream>>>(x, Bp1, gbuf, nRowTiles);
  hipMemsetAsync(agg, 0, (size_t)N * 64 * sizeof(float), stream);
  GCN_scatter_kernel<16, 4><<<sc64, 256, 0, stream>>>(rc, nrm,
                                                      (const float4*)gbuf, agg, E);
  GCN_finalize_kernel<16, 4><<<fin64, 256, 0, stream>>>((const float4*)agg,
                                                        (const float4*)gbuf, dinv,
                                                        (const float4*)b1,
                                                        (float4*)pen, N, 1);

  // --- layer 2: pen = relu(Agg(h1@W2) + b2) (in place) ---------------------
  GCN_gemm_wmma_f16<64><<<g64, 128, 0, stream>>>(pen, Bp2, gbuf, nRowTiles);
  hipMemsetAsync(agg, 0, (size_t)N * 64 * sizeof(float), stream);
  GCN_scatter_kernel<16, 4><<<sc64, 256, 0, stream>>>(rc, nrm,
                                                      (const float4*)gbuf, agg, E);
  GCN_finalize_kernel<16, 4><<<fin64, 256, 0, stream>>>((const float4*)agg,
                                                        (const float4*)gbuf, dinv,
                                                        (const float4*)b2,
                                                        (float4*)pen, N, 1);

  // --- layer 3: out = Agg(pen@Wo) + bo (no relu) ---------------------------
  GCN_gemm_wmma_f16<16><<<g16, 128, 0, stream>>>(pen, Bpo, gbuf, nRowTiles);
  hipMemsetAsync(agg, 0, (size_t)N * 16 * sizeof(float), stream);
  GCN_scatter_kernel<4, 2><<<sc16, 256, 0, stream>>>(rc, nrm,
                                                     (const float4*)gbuf, agg, E);
  GCN_finalize_kernel<4, 2><<<fin16, 256, 0, stream>>>((const float4*)agg,
                                                       (const float4*)gbuf, dinv,
                                                       (const float4*)bo,
                                                       (float4*)out0, N, 0);
}
